// MultiHeadedSelfAttention_35433480192334
// MI455X (gfx1250) — matrix operand order repaired and checked by reading, not compiled, not run on hardware
//
#include <hip/hip_runtime.h>
#include <hip/hip_bf16.h>
#include <stdint.h>

// FAVOR+ (Performer) attention, chunked-parallel reformulation for gfx1250.
// All GEMMs run on v_wmma_f32_16x16x32_f16 (f16 in, f32 accumulate).
// B fragments are hoisted ahead of the WMMA burst so loads clause together
// and s_wait_loadcnt can drain incrementally instead of fully serializing.

typedef __attribute__((ext_vector_type(16))) _Float16 v16h;
typedef __attribute__((ext_vector_type(8)))  float    v8f;

#define BATCH  4
#define SEQ    2048
#define DIM    256
#define ROWS   (BATCH * SEQ)   // 8192
#define CHUNK  64
#define NCHUNK (SEQ / CHUNK)   // 32

__device__ __forceinline__ v8f wmma_f16f32(v16h a, v16h b, v8f c) {
  // 8 args: (neg_a, A, neg_b, B, c_mod, C, reuse_a, reuse_b)
  return __builtin_amdgcn_wmma_f32_16x16x32_f16(false, a, false, b, (short)0, c,
                                                false, false);
}

// ---- fragment loaders --------------------------------------------------
// A (16x32 f16, MxK): lane L -> row M = L%16, khalf = L/16,
// vgpr v packs K = 16*khalf + 2*v + {0,1} (matches load_b_h).
__device__ __forceinline__ v16h load_a_h(const _Float16* src, int ld, int row0, int k0) {
  int lane = threadIdx.x & 31;
  int rm = lane & 15, kh = lane >> 4;
  const _Float16* p = src + (size_t)(row0 + rm) * ld + k0 + 16 * kh;
  union { v16h h; unsigned int u[8]; } fr;
#pragma unroll
  for (int v = 0; v < 8; ++v) {
    int kk = 2 * v;
    fr.u[v] = *(const unsigned int*)(p + kk);   // aligned pair of halves
  }
  return fr.h;
}

// A fragment from an f32 row-major source, converting to f16 on the fly.
__device__ __forceinline__ v16h load_a_f(const float* src, int ld, int row0, int k0) {
  int lane = threadIdx.x & 31;
  int rm = lane & 15, kh = lane >> 4;
  const float* p = src + (size_t)(row0 + rm) * ld + k0 + 16 * kh;
  union { v16h h; unsigned int u[8]; } fr;
#pragma unroll
  for (int v = 0; v < 8; ++v) {
    int kk = 2 * v;
    float2 f = *(const float2*)(p + kk);
    union { _Float16 h2[2]; unsigned int u; } pk;
    pk.h2[0] = (_Float16)f.x; pk.h2[1] = (_Float16)f.y;
    fr.u[v] = pk.u;
  }
  return fr.h;
}

// B (32x16 f16, KxN) stored as [N, K] row-major (K contiguous):
// lane L -> col N = L%16, K = 16*(L/16) + e  (16 contiguous halves = 32B).
__device__ __forceinline__ v16h load_b_h(const _Float16* src, int ld, int n0, int k0) {
  int lane = threadIdx.x & 31;
  int n = lane & 15, kh = lane >> 4;
  return *(const v16h*)(src + (size_t)(n0 + n) * ld + k0 + 16 * kh);
}

// ---- kernel 0: weight prep (f32 -> f16, rfs transposed) ----------------
__global__ void k0_prep(const float* __restrict__ Wq, const float* __restrict__ Wk,
                        const float* __restrict__ Wv, const float* __restrict__ rfs,
                        _Float16* __restrict__ WH, _Float16* __restrict__ rfsT) {
  int idx = blockIdx.x * 256 + threadIdx.x;       // grid 1024 * 256 = 4*65536
  int mat = idx >> 16, r = (idx >> 8) & 255, c = idx & 255;
  if (mat < 3) {
    const float* W = (mat == 0) ? Wq : ((mat == 1) ? Wk : Wv);
    WH[(size_t)mat * 65536 + r * 256 + c] = (_Float16)W[r * 256 + c];
  } else {
    rfsT[(size_t)c * 256 + r] = (_Float16)rfs[r * 256 + c];  // rfsT[m][l]
  }
}

// ---- kernel 1: Q/K/V projection GEMMs ----------------------------------
// grid (128, 3): x = 64-row tile, y selects q/k/v.  q,k scaled by D^-0.25.
// V is written TRANSPOSED per batch: VT[b][d][s] (t contiguous).
__global__ void k1_qkv(const float* __restrict__ x, const _Float16* __restrict__ WH,
                       const float* __restrict__ bq, const float* __restrict__ bk,
                       const float* __restrict__ bv,
                       _Float16* __restrict__ Qs, _Float16* __restrict__ Ks,
                       _Float16* __restrict__ VT) {
  int which = blockIdx.y;
  const _Float16* W = WH + (size_t)which * 65536;
  const float* bias = (which == 0) ? bq : ((which == 1) ? bk : bv);
  int wave = threadIdx.x >> 5, lane = threadIdx.x & 31;
  int r = wave >> 1, ch = wave & 1;
  int rowBase = blockIdx.x * 64 + 16 * r;
  v8f acc[8];
#pragma unroll
  for (int j = 0; j < 8; ++j) acc[j] = {};
  for (int k0 = 0; k0 < DIM; k0 += 32) {
    __builtin_prefetch(x + (size_t)rowBase * DIM + k0 + 64, 0, 1);
    v16h a = load_a_f(x, DIM, rowBase, k0);
    v16h bf[8];
#pragma unroll
    for (int j = 0; j < 8; ++j) bf[j] = load_b_h(W, DIM, 128 * ch + 16 * j, k0);
#pragma unroll
    for (int j = 0; j < 8; ++j) acc[j] = wmma_f16f32(a, bf[j], acc[j]);
  }
  int kh = lane >> 4, n = lane & 15;
#pragma unroll
  for (int j = 0; j < 8; ++j) {
    int col = 128 * ch + 16 * j + n;
    float bval = bias[col];
#pragma unroll
    for (int i = 0; i < 8; ++i) {
      int row = rowBase + i + 8 * kh;
      float val = acc[j][i] + bval;
      if (which == 0) {
        Qs[(size_t)row * DIM + col] = (_Float16)(val * 0.25f);
      } else if (which == 1) {
        Ks[(size_t)row * DIM + col] = (_Float16)(val * 0.25f);
      } else {
        int bb = row >> 11, sl = row & 2047;
        VT[(((size_t)(bb * DIM + col)) << 11) + sl] = (_Float16)val;
      }
    }
  }
}

// ---- kernel 1b: row norms 0.5*||q||^2 ----------------------------------
__global__ void k1_rn(const _Float16* __restrict__ Qs, const _Float16* __restrict__ Ks,
                      float* __restrict__ rnQ, float* __restrict__ rnK) {
  const _Float16* src = blockIdx.y ? Ks : Qs;
  float* rn = blockIdx.y ? rnK : rnQ;
  int wave = threadIdx.x >> 5, lane = threadIdx.x & 31;
  int row = blockIdx.x * 8 + wave;
  const _Float16* p = src + (size_t)row * DIM + lane * 8;
  float s = 0.f;
#pragma unroll
  for (int t = 0; t < 8; ++t) { float h = (float)p[t]; s += h * h; }
#pragma unroll
  for (int off = 16; off > 0; off >>= 1) s += __shfl_xor(s, off, 32);
  if (lane == 0) rn[row] = 0.5f * s;
}

// ---- kernel 2: random-feature maps  QF/KF = exp(Qs@rfs - rn) -----------
// grid (128, 2): y=0 -> QF, y=1 -> KF (+ transposed copy KFT[b][m][s]).
__global__ void k2_rf(const _Float16* __restrict__ Qs, const _Float16* __restrict__ Ks,
                      const _Float16* __restrict__ rfsT,
                      const float* __restrict__ rnQ, const float* __restrict__ rnK,
                      _Float16* __restrict__ QF, _Float16* __restrict__ KF,
                      _Float16* __restrict__ KFT) {
  bool isK = (blockIdx.y != 0);
  const _Float16* src = isK ? Ks : Qs;
  const float* rn = isK ? rnK : rnQ;
  _Float16* dst = isK ? KF : QF;
  int wave = threadIdx.x >> 5, lane = threadIdx.x & 31;
  int r = wave >> 1, ch = wave & 1;
  int rowBase = blockIdx.x * 64 + 16 * r;
  v8f acc[8];
#pragma unroll
  for (int j = 0; j < 8; ++j) acc[j] = {};
  for (int k0 = 0; k0 < DIM; k0 += 32) {
    v16h a = load_a_h(src, DIM, rowBase, k0);
    v16h bf[8];
#pragma unroll
    for (int j = 0; j < 8; ++j) bf[j] = load_b_h(rfsT, DIM, 128 * ch + 16 * j, k0);
#pragma unroll
    for (int j = 0; j < 8; ++j) acc[j] = wmma_f16f32(a, bf[j], acc[j]);
  }
  int kh = lane >> 4, n = lane & 15;
#pragma unroll
  for (int j = 0; j < 8; ++j) {
    int col = 128 * ch + 16 * j + n;
#pragma unroll
    for (int i = 0; i < 8; ++i) {
      int row = rowBase + i + 8 * kh;
      float e = __expf(acc[j][i] - rn[row]);
      dst[(size_t)row * DIM + col] = (_Float16)e;
      if (isK) {
        int bb = row >> 11, sl = row & 2047;
        KFT[(((size_t)(bb * DIM + col)) << 11) + sl] = (_Float16)e;
      }
    }
  }
}

// ---- kernel 3a: per-chunk states  KVT_c[d][m] = sum_t V[t,d]*KF[t,m] ----
// grid (4 dBlocks, 32 chunks, 4 batches). A = VT, B = KFT (both t-contig).
__global__ void k3a_state(const _Float16* __restrict__ VT, const _Float16* __restrict__ KFT,
                          _Float16* __restrict__ KVC) {
  int b = blockIdx.z, c = blockIdx.y, d0 = blockIdx.x * 64;
  int wave = threadIdx.x >> 5, lane = threadIdx.x & 31;
  int r = wave >> 1, ch = wave & 1;
  int rowBase = d0 + 16 * r;
  const _Float16* vtb = VT + ((size_t)b * DIM << 11);
  const _Float16* kftb = KFT + ((size_t)b * DIM << 11);
  int tb = c * CHUNK;
  v8f acc[8];
#pragma unroll
  for (int j = 0; j < 8; ++j) acc[j] = {};
#pragma unroll
  for (int k0 = 0; k0 < CHUNK; k0 += 32) {
    v16h a = load_a_h(vtb, SEQ, rowBase, tb + k0);
    v16h bf[8];
#pragma unroll
    for (int j = 0; j < 8; ++j) bf[j] = load_b_h(kftb, SEQ, 128 * ch + 16 * j, tb + k0);
#pragma unroll
    for (int j = 0; j < 8; ++j) acc[j] = wmma_f16f32(a, bf[j], acc[j]);
  }
  int kh = lane >> 4, n = lane & 15;
  size_t base = (size_t)(b * NCHUNK + c) * 65536;
#pragma unroll
  for (int j = 0; j < 8; ++j) {
    int m = 128 * ch + 16 * j + n;
#pragma unroll
    for (int i = 0; i < 8; ++i) {
      int d = rowBase + i + 8 * kh;
      KVC[base + (size_t)d * DIM + m] = (_Float16)acc[j][i];
    }
  }
}

// per-chunk key sums  Ksum_c[m] = sum_t KF[t,m]
__global__ void k3a_ksum(const _Float16* __restrict__ KFT, float* __restrict__ KSC) {
  int c = blockIdx.x, b = blockIdx.y, m = threadIdx.x;
  const _Float16* p = KFT + (((size_t)(b * DIM + m)) << 11) + c * CHUNK;
  float s = 0.f;
#pragma unroll
  for (int t = 0; t < CHUNK; ++t) s += (float)p[t];
  KSC[(size_t)(b * NCHUNK + c) * DIM + m] = s;
}

// ---- kernel 3b: exclusive prefix over chunk states (f32 accum) ---------
// grid (8 dSlices, 4 batches); each thread owns a 32-float strip in VGPRs.
__global__ void k3b_prefix(const _Float16* __restrict__ KVC, _Float16* __restrict__ KVP) {
  int b = blockIdx.y;
  int d = blockIdx.x * 32 + (threadIdx.x >> 3);
  int mb = (threadIdx.x & 7) * 32;
  float acc[32];
#pragma unroll
  for (int i = 0; i < 32; ++i) acc[i] = 0.f;
  size_t rowoff = (size_t)d * DIM + mb;
  for (int c = 0; c < NCHUNK; ++c) {
    size_t base = (size_t)(b * NCHUNK + c) * 65536 + rowoff;
#pragma unroll
    for (int i = 0; i < 32; ++i) KVP[base + i] = (_Float16)acc[i];
#pragma unroll
    for (int i = 0; i < 32; ++i) acc[i] += (float)KVC[base + i];
  }
}

__global__ void k3b_ksum(const float* __restrict__ KSC, float* __restrict__ KSP) {
  int b = blockIdx.x, m = threadIdx.x;
  float a = 0.f;
  for (int c = 0; c < NCHUNK; ++c) {
    size_t idx = (size_t)(b * NCHUNK + c) * DIM + m;
    KSP[idx] = a;
    a += KSC[idx];
  }
}

// ---- kernel 3c: per-chunk output ---------------------------------------
// O = [tril(QF_c@KF_c^T)]@V_c + QF_c@KVprefix ; out = O / (den + 1e-16)
// grid (32 chunks, 4 batches), 256 threads.
__global__ void k3c_out(const _Float16* __restrict__ QF, const _Float16* __restrict__ KF,
                        const _Float16* __restrict__ VT, const _Float16* __restrict__ KVP,
                        const float* __restrict__ KSP, float* __restrict__ out) {
  int c = blockIdx.x, b = blockIdx.y;
  int gr = b * SEQ + c * CHUNK;                 // global row of chunk start
  __shared__ _Float16 Sh[CHUNK * CHUNK];        // masked scores, f16
  __shared__ float den_part[CHUNK * 4];         // intra-chunk den partials
  __shared__ float den_p2[CHUNK * 4];           // inter-chunk den partials
  __shared__ float den_s[CHUNK];
  int tid = threadIdx.x;
  int wave = tid >> 5, lane = tid & 31;
  int kh = lane >> 4, n = lane & 15;

  // -- step 1: scores S = QF_c @ KF_c^T, causal mask (inclusive), stash --
  {
    int r = wave >> 1, cp = wave & 1;           // wave owns tiles (r, 2cp..2cp+1)
    v8f sa[2]; sa[0] = {}; sa[1] = {};
    for (int k0 = 0; k0 < DIM; k0 += 32) {
      v16h a = load_a_h(QF, DIM, gr + 16 * r, k0);
      v16h bf[2];
#pragma unroll
      for (int jj = 0; jj < 2; ++jj)
        bf[jj] = load_b_h(KF + (size_t)gr * DIM, DIM, 16 * (2 * cp + jj), k0);
#pragma unroll
      for (int jj = 0; jj < 2; ++jj) sa[jj] = wmma_f16f32(a, bf[jj], sa[jj]);
    }
#pragma unroll
    for (int jj = 0; jj < 2; ++jj) {
      int cb = 2 * cp + jj;
#pragma unroll
      for (int i = 0; i < 8; ++i) {
        int t = 16 * r + i + 8 * kh;
        int tp = 16 * cb + n;
        float val = (tp <= t) ? sa[jj][i] : 0.f;
        Sh[t * CHUNK + tp] = (_Float16)val;
        float p = val;
#pragma unroll
        for (int off = 1; off < 16; off <<= 1) p += __shfl_xor(p, off, 32);
        if (n == 0) den_part[t * 4 + cb] = p;   // unique (t, cb) writer
      }
    }
  }
  __syncthreads();

  // -- step 2: den_inter[t] = QF[t,:] . KsumPrefix --
  {
    int t = tid >> 2, part = tid & 3;
    const _Float16* q = QF + (size_t)(gr + t) * DIM + part * 64;
    const float* ks = KSP + (size_t)(b * NCHUNK + c) * DIM + part * 64;
    float s = 0.f;
#pragma unroll
    for (int m = 0; m < 64; ++m) s += (float)q[m] * ks[m];
    den_p2[t * 4 + part] = s;
  }
  __syncthreads();
  if (tid < CHUNK) {
    den_s[tid] = den_part[tid * 4] + den_part[tid * 4 + 1] + den_part[tid * 4 + 2] +
                 den_part[tid * 4 + 3] + den_p2[tid * 4] + den_p2[tid * 4 + 1] +
                 den_p2[tid * 4 + 2] + den_p2[tid * 4 + 3];
  }
  __syncthreads();

  // -- step 3: O = QF@KVP (K=256)  +  S@V (K=64); divide by den --
  {
    int r = wave >> 1, ch = wave & 1;
    v8f acc[8];
#pragma unroll
    for (int j = 0; j < 8; ++j) acc[j] = {};
    const _Float16* kvp = KVP + (size_t)(b * NCHUNK + c) * 65536;
    for (int k0 = 0; k0 < DIM; k0 += 32) {
      v16h a = load_a_h(QF, DIM, gr + 16 * r, k0);
      v16h bf[8];
#pragma unroll
      for (int j = 0; j < 8; ++j) bf[j] = load_b_h(kvp, DIM, 128 * ch + 16 * j, k0);
#pragma unroll
      for (int j = 0; j < 8; ++j) acc[j] = wmma_f16f32(a, bf[j], acc[j]);
    }
    const _Float16* vtb = VT + ((size_t)b * DIM << 11);
#pragma unroll
    for (int k0 = 0; k0 < CHUNK; k0 += 32) {
      v16h a = load_a_h(Sh, CHUNK, 16 * r, k0);
      v16h bf[8];
#pragma unroll
      for (int j = 0; j < 8; ++j)
        bf[j] = load_b_h(vtb, SEQ, 128 * ch + 16 * j, c * CHUNK + k0);
#pragma unroll
      for (int j = 0; j < 8; ++j) acc[j] = wmma_f16f32(a, bf[j], acc[j]);
    }
#pragma unroll
    for (int j = 0; j < 8; ++j) {
      int d = 128 * ch + 16 * j + n;
#pragma unroll
      for (int i = 0; i < 8; ++i) {
        int t = 16 * r + i + 8 * kh;
        float o = acc[j][i] / (den_s[t] + 1e-16f);
        out[(size_t)(gr + t) * DIM + d] = o;
      }
    }
  }
}

// ---- workspace layout (bytes) ------------------------------------------
#define WS_RFST  ((size_t)0)               // 256*256*2        = 128 KB
#define WS_WH    ((size_t)131072)          // 3*256*256*2      = 384 KB
#define WS_QS    ((size_t)524288)          // 8192*256*2       = 4 MB
#define WS_KS    ((size_t)4718592)
#define WS_VT    ((size_t)8912896)
#define WS_QF    ((size_t)13107200)
#define WS_KF    ((size_t)17301504)
#define WS_KFT   ((size_t)21495808)
#define WS_RNQ   ((size_t)25690112)        // 8192*4
#define WS_RNK   ((size_t)25722880)
#define WS_KVC   ((size_t)25755648)        // 4*32*65536*2     = 16 MB
#define WS_KVP   ((size_t)42532864)        // 16 MB
#define WS_KSC   ((size_t)59310080)        // 4*32*256*4
#define WS_KSP   ((size_t)59441152)        // total ~56.8 MB

extern "C" void kernel_launch(void* const* d_in, const int* in_sizes, int n_in,
                              void* d_out, int out_size, void* d_ws, size_t ws_size,
                              hipStream_t stream) {
  (void)in_sizes; (void)n_in; (void)out_size; (void)ws_size;
  const float* x   = (const float*)d_in[0];
  // d_in[1] = mask (unused on the favor+ path)
  const float* Wq  = (const float*)d_in[2];
  const float* bq  = (const float*)d_in[3];
  const float* Wk  = (const float*)d_in[4];
  const float* bk  = (const float*)d_in[5];
  const float* Wv  = (const float*)d_in[6];
  const float* bv  = (const float*)d_in[7];
  const float* rfs = (const float*)d_in[8];

  char* ws = (char*)d_ws;
  _Float16* rfsT = (_Float16*)(ws + WS_RFST);
  _Float16* WH   = (_Float16*)(ws + WS_WH);
  _Float16* Qs   = (_Float16*)(ws + WS_QS);
  _Float16* Ks   = (_Float16*)(ws + WS_KS);
  _Float16* VT   = (_Float16*)(ws + WS_VT);
  _Float16* QF   = (_Float16*)(ws + WS_QF);
  _Float16* KF   = (_Float16*)(ws + WS_KF);
  _Float16* KFT  = (_Float16*)(ws + WS_KFT);
  float*    rnQ  = (float*)(ws + WS_RNQ);
  float*    rnK  = (float*)(ws + WS_RNK);
  _Float16* KVC  = (_Float16*)(ws + WS_KVC);
  _Float16* KVP  = (_Float16*)(ws + WS_KVP);
  float*    KSC  = (float*)(ws + WS_KSC);
  float*    KSP  = (float*)(ws + WS_KSP);

  k0_prep   <<<dim3(1024),      256, 0, stream>>>(Wq, Wk, Wv, rfs, WH, rfsT);
  k1_qkv    <<<dim3(128, 3),    256, 0, stream>>>(x, WH, bq, bk, bv, Qs, Ks, VT);
  k1_rn     <<<dim3(1024, 2),   256, 0, stream>>>(Qs, Ks, rnQ, rnK);
  k2_rf     <<<dim3(128, 2),    256, 0, stream>>>(Qs, Ks, rfsT, rnQ, rnK, QF, KF, KFT);
  k3a_state <<<dim3(4, 32, 4),  256, 0, stream>>>(VT, KFT, KVC);
  k3a_ksum  <<<dim3(32, 4),     256, 0, stream>>>(KFT, KSC);
  k3b_prefix<<<dim3(8, 4),      256, 0, stream>>>(KVC, KVP);
  k3b_ksum  <<<dim3(4),         256, 0, stream>>>(KSC, KSP);
  k3c_out   <<<dim3(32, 4),     256, 0, stream>>>(QF, KF, VT, KVP, KSP, (float*)d_out);
}
